// PrototypicalNet_60361470378722
// MI455X (gfx1250) — compile-verified, matrix operand order checked by reading, not run on hardware
//
#include <hip/hip_runtime.h>

// ---------------------------------------------------------------------------
// PrototypicalNet forward for MI455X (gfx1250, wave32, WMMA).
//   1) transpose+convert W1,W2 (fp32 [K][N]) -> bf16 [N][K] in workspace
//   2) GEMM1 (bf16 WMMA, f32 acc): H = relu(X @ W1 + b1)   -> bf16
//   3) GEMM2 (bf16 WMMA, f32 acc): E = H @ W2 + b2         -> f32
//   4) prototypes: mean over 32 shots per class            -> f32 [64][512]
//   5) distances: -sqrt(sum((p - q + eps)^2))              -> f32 [16384][64]
// GEMMs: 128x128 block tile, BK=64, double-buffered LDS, async global->LDS
// (ASYNCcnt) for bf16-resident operands, fp32->bf16 conversion in-register
// for the layer-1 activations.
// ---------------------------------------------------------------------------

typedef __attribute__((ext_vector_type(16))) __bf16 v16bf;
typedef __attribute__((ext_vector_type(8)))  float  v8f;

#define PN_EPS 1e-6f
#define BM 128
#define BN 128
#define BK 64
#define LDSW 36u              // dwords per LDS row: 32 used (64 bf16) + 4 pad
#define TILE_DW (BM * LDSW)   // 4608 dwords per tile buffer

#if __has_builtin(__builtin_amdgcn_global_load_async_to_lds_b128)
#define HAS_ASYNC_LDS 1
#else
#define HAS_ASYNC_LDS 0
#endif

// Builtin parameter types (from hipcc diagnostic): pointer to int
// __attribute__((__vector_size__(16))) in AS(1) (printed "__device__"),
// non-const. Mirror for the LDS side in AS(3).
typedef __attribute__((__vector_size__(4 * sizeof(int)))) int v4i_t;
typedef __attribute__((address_space(1))) v4i_t gv4i;
typedef __attribute__((address_space(3))) v4i_t lv4i;

__device__ __forceinline__ unsigned short f2bf(float f) {
  unsigned int u = __float_as_uint(f);
  u += 0x7FFFu + ((u >> 16) & 1u);   // round-to-nearest-even
  return (unsigned short)(u >> 16);
}

__device__ __forceinline__ unsigned int pk2bf(float a, float b) {
  unsigned int ua = __float_as_uint(a);
  unsigned int ub = __float_as_uint(b);
  ua += 0x7FFFu + ((ua >> 16) & 1u);
  ub += 0x7FFFu + ((ub >> 16) & 1u);
  return (ua >> 16) | (ub & 0xFFFF0000u);
}

// 16-byte global -> LDS copy; async (ASYNCcnt) when the gfx1250 builtin exists.
__device__ __forceinline__ void cp16_g2l(const void* g, unsigned int* l) {
#if HAS_ASYNC_LDS
  // generic->AS1/AS3 via integer detour: generic LDS addr low 32 bits are the
  // LDS offset (aperture rule), generic global addr == 64-bit VA.
  __builtin_amdgcn_global_load_async_to_lds_b128(
      (gv4i*)(unsigned long long)g,
      (lv4i*)(unsigned int)(unsigned long long)l,
      0, 0);
#else
  *(uint4*)l = *(const uint4*)g;
#endif
}

template<int N> __device__ __forceinline__ void wait_async() {
#if HAS_ASYNC_LDS
#if __has_builtin(__builtin_amdgcn_s_wait_asynccnt)
  __builtin_amdgcn_s_wait_asynccnt(N);
#else
  asm volatile("s_wait_asynccnt %0" :: "i"(N) : "memory");
#endif
#endif
}

union FragU { uint4 u[2]; v16bf v; };

// ---------------------------------------------------------------------------
// Weight transpose + fp32->bf16 convert: W [K][N] f32 -> Wt [N][K] bf16
// ---------------------------------------------------------------------------
__global__ __launch_bounds__(256)
void transpose_cvt_bf16(const float* __restrict__ W,
                        unsigned short* __restrict__ Wt,
                        int K, int N) {
  __shared__ float tile[32][33];
  const int kb = blockIdx.x * 32;
  const int nb = blockIdx.y * 32;
  const int tx = threadIdx.x;   // 0..31
  const int ty = threadIdx.y;   // 0..7
  #pragma unroll
  for (int i = 0; i < 32; i += 8)
    tile[ty + i][tx] = W[(size_t)(kb + ty + i) * N + nb + tx];
  __syncthreads();
  #pragma unroll
  for (int i = 0; i < 32; i += 8)
    Wt[(size_t)(nb + ty + i) * K + kb + tx] = f2bf(tile[tx][ty + i]);
}

// ---------------------------------------------------------------------------
// Tiled bf16 WMMA GEMM: C[M][N] = act(A[M][K] @ Bt[N][K]^T + bias)
// Block 256 thr (8 waves); block tile 128x128; BK=64 (two WMMA k-halves);
// double-buffered LDS (72 KB); wave tile 32x64 = 2x4 fragments.
// ---------------------------------------------------------------------------
template<bool A_IS_F32, bool RELU, bool OUT_BF16>
__global__ __launch_bounds__(256)
void gemm_bf16_wmma(const void* __restrict__ Aany,
                    const unsigned short* __restrict__ Bt,
                    const float* __restrict__ bias,
                    void* __restrict__ Cany,
                    int M, int N, int K) {
  __shared__ unsigned int lds[4 * TILE_DW];  // [buf][A|B], 72 KB

  const int t    = threadIdx.x;
  const int lane = t & 31;
  const int wid  = t >> 5;
  const int wm   = (wid & 3) * 32;
  const int wn   = (wid >> 2) * 64;
  const int lrow = lane & 15;
  const int hi16 = lane >> 4;

  const int m0 = blockIdx.x * BM;
  const int n0 = blockIdx.y * BN;
  const int nk = K / BK;

  v8f acc[2][4] = {};

  // ---- tile staging (stage ks into buffer buf) ----
  auto stage_tiles = [&](int ks, int buf) {
    const int k0 = ks * BK;
    unsigned int* pA = lds + buf * 2 * TILE_DW;
    unsigned int* pB = pA + TILE_DW;
    if (A_IS_F32) {
      const float* A = (const float*)Aany;
      #pragma unroll
      for (int p = 0; p < 8; ++p) {
        int row = (t >> 4) + p * 16;
        int c4  = (t & 15) << 2;
        float4 v = *(const float4*)(A + (size_t)(m0 + row) * K + k0 + c4);
        uint2 w;
        w.x = pk2bf(v.x, v.y);
        w.y = pk2bf(v.z, v.w);
        *(uint2*)(pA + row * LDSW + (c4 >> 1)) = w;
      }
    } else {
      const unsigned short* A = (const unsigned short*)Aany;
      #pragma unroll
      for (int p = 0; p < 4; ++p) {
        int i = t + p * 256;
        int row = i >> 3;
        int q   = i & 7;
        cp16_g2l(A + (size_t)(m0 + row) * K + k0 + q * 8,
                 pA + row * LDSW + q * 4);
      }
    }
    #pragma unroll
    for (int p = 0; p < 4; ++p) {
      int i = t + p * 256;
      int row = i >> 3;
      int q   = i & 7;
      cp16_g2l(Bt + (size_t)(n0 + row) * K + k0 + q * 8,
               pB + row * LDSW + q * 4);
    }
  };

  stage_tiles(0, 0);

  for (int ks = 0; ks < nk; ++ks) {
    const int buf = ks & 1;
    const bool has_next = (ks + 1) < nk;
    if (has_next) stage_tiles(ks + 1, buf ^ 1);

    // current stage's async copies are complete once only the next stage's
    // (issued above, in-order completion) remain outstanding
    if (has_next) { if (A_IS_F32) wait_async<4>(); else wait_async<8>(); }
    else          wait_async<0>();
    __syncthreads();

    const unsigned int* pA = lds + buf * 2 * TILE_DW;
    const unsigned int* pB = pA + TILE_DW;

    #pragma unroll
    for (int kk = 0; kk < 2; ++kk) {
      const int kd = kk * 16;  // dword offset of k-half within row
      v16bf af[2], bfrag[4];
      #pragma unroll
      for (int m = 0; m < 2; ++m) {
        int r = wm + m * 16 + lrow;
        FragU f;
        f.u[0] = *(const uint4*)(pA + r * LDSW + kd + hi16 * 4);
        f.u[1] = *(const uint4*)(pA + r * LDSW + kd + hi16 * 4 + 8);
        af[m] = f.v;
      }
      #pragma unroll
      for (int n = 0; n < 4; ++n) {
        int r = wn + n * 16 + lrow;
        FragU f;
        f.u[0] = *(const uint4*)(pB + r * LDSW + kd + hi16 * 4);
        f.u[1] = *(const uint4*)(pB + r * LDSW + kd + hi16 * 4 + 8);
        bfrag[n] = f.v;
      }
      #pragma unroll
      for (int m = 0; m < 2; ++m)
        #pragma unroll
        for (int n = 0; n < 4; ++n)
          acc[m][n] = __builtin_amdgcn_wmma_f32_16x16x32_bf16(
              false, af[m], false, bfrag[n], (short)0, acc[m][n], false, false);
    }
    __syncthreads();
  }

  // ---- epilogue: bias (+relu). C/D layout: col = lane%16, M = v + 8*hi16
  #pragma unroll
  for (int n = 0; n < 4; ++n) {
    int col = n0 + wn + n * 16 + lrow;
    float bv = bias[col];
    #pragma unroll
    for (int m = 0; m < 2; ++m) {
      #pragma unroll
      for (int v = 0; v < 8; ++v) {
        int row = m0 + wm + m * 16 + (hi16 << 3) + v;
        float x = acc[m][n][v] + bv;
        if (RELU) x = fmaxf(x, 0.0f);
        if (OUT_BF16)
          ((unsigned short*)Cany)[(size_t)row * N + col] = f2bf(x);
        else
          ((float*)Cany)[(size_t)row * N + col] = x;
      }
    }
  }
}

// ---------------------------------------------------------------------------
// Prototypes: P[c][d] = mean over 32 shots of Es[(c*32+s)][d]   (64x512)
// ---------------------------------------------------------------------------
__global__ __launch_bounds__(256)
void proto_mean(const float* __restrict__ Es, float* __restrict__ P) {
  int id = blockIdx.x * 256 + threadIdx.x;   // 32768 total
  int c = id >> 9;
  int d = id & 511;
  float s = 0.0f;
  #pragma unroll 8
  for (int j = 0; j < 32; ++j)
    s += Es[(size_t)((c * 32 + j) << 9) + d];
  P[id] = s * (1.0f / 32.0f);
}

// ---------------------------------------------------------------------------
// Distances: out[q][c] = -sqrt(sum_d (P[c][d] - Eq[q][d] + eps)^2)
// One wave per query; lanes across D=512; shuffle reduce per class.
// ---------------------------------------------------------------------------
__global__ __launch_bounds__(256)
void dist_kernel(const float* __restrict__ Eq, const float* __restrict__ P,
                 float* __restrict__ out) {
  const int lane = threadIdx.x & 31;
  const int wid  = threadIdx.x >> 5;
  const int q    = blockIdx.x * 8 + wid;   // 2048 blocks * 8 waves = 16384
  const float* qr = Eq + (size_t)q * 512;

  float4 qv[4];
  #pragma unroll
  for (int j = 0; j < 4; ++j)
    qv[j] = *(const float4*)(qr + lane * 4 + j * 128);

  for (int c = 0; c < 64; ++c) {
    const float* pr = P + c * 512;
    float s = 0.0f;
    #pragma unroll
    for (int j = 0; j < 4; ++j) {
      float4 pv = *(const float4*)(pr + lane * 4 + j * 128);
      float d0 = pv.x - qv[j].x + PN_EPS;
      float d1 = pv.y - qv[j].y + PN_EPS;
      float d2 = pv.z - qv[j].z + PN_EPS;
      float d3 = pv.w - qv[j].w + PN_EPS;
      s = fmaf(d0, d0, s);
      s = fmaf(d1, d1, s);
      s = fmaf(d2, d2, s);
      s = fmaf(d3, d3, s);
    }
    #pragma unroll
    for (int off = 16; off > 0; off >>= 1)
      s += __shfl_xor(s, off, 32);
    if (lane == 0)
      out[(size_t)q * 64 + c] = -sqrtf(s);
  }
}

// ---------------------------------------------------------------------------
extern "C" void kernel_launch(void* const* d_in, const int* in_sizes, int n_in,
                              void* d_out, int out_size, void* d_ws, size_t ws_size,
                              hipStream_t stream) {
  (void)in_sizes; (void)n_in; (void)out_size; (void)ws_size;

  const float* Xs = (const float*)d_in[0];  // [2048, 2048]
  const float* Xq = (const float*)d_in[1];  // [16384, 2048]
  const float* W1 = (const float*)d_in[2];  // [2048, 1024]
  const float* b1 = (const float*)d_in[3];  // [1024]
  const float* W2 = (const float*)d_in[4];  // [1024, 512]
  const float* b2 = (const float*)d_in[5];  // [512]
  float* out = (float*)d_out;               // [16384, 64]

  // Workspace layout (bytes), ~77.1 MB total
  char* ws = (char*)d_ws;
  unsigned short* W1T = (unsigned short*)(ws + 0);          //  4 MB  [1024][2048] bf16
  unsigned short* W2T = (unsigned short*)(ws + 4194304);    //  1 MB  [512][1024]  bf16
  unsigned short* Hs  = (unsigned short*)(ws + 5242880);    //  4 MB  [2048][1024] bf16
  unsigned short* Hq  = (unsigned short*)(ws + 9437184);    // 32 MB  [16384][1024] bf16
  float*          Es  = (float*)(ws + 42991616);            //  4 MB  [2048][512]  f32
  float*          Eq  = (float*)(ws + 47185920);            // 32 MB  [16384][512] f32
  float*          P   = (float*)(ws + 80740352);            // 128 KB [64][512]    f32

  dim3 tb(32, 8);
  transpose_cvt_bf16<<<dim3(64, 32), tb, 0, stream>>>(W1, W1T, 2048, 1024);
  transpose_cvt_bf16<<<dim3(32, 16), tb, 0, stream>>>(W2, W2T, 1024, 512);

  // Layer 1: relu(X @ W1 + b1) -> bf16
  gemm_bf16_wmma<true, true, true><<<dim3(16, 8),  256, 0, stream>>>(Xs, W1T, b1, Hs, 2048, 1024, 2048);
  gemm_bf16_wmma<true, true, true><<<dim3(128, 8), 256, 0, stream>>>(Xq, W1T, b1, Hq, 16384, 1024, 2048);

  // Layer 2: H @ W2 + b2 -> f32 embeddings
  gemm_bf16_wmma<false, false, false><<<dim3(16, 4),  256, 0, stream>>>(Hs, W2T, b2, Es, 2048, 512, 1024);
  gemm_bf16_wmma<false, false, false><<<dim3(128, 4), 256, 0, stream>>>(Hq, W2T, b2, Eq, 16384, 512, 1024);

  proto_mean<<<128, 256, 0, stream>>>(Es, P);
  dist_kernel<<<2048, 256, 0, stream>>>(Eq, P, out);
}
